// VectorQuantizerEMA_90005334655877
// MI455X (gfx1250) — compile-verified, matrix operand order checked by reading, not run on hardware
//
#include <hip/hip_runtime.h>
#include <hip/hip_bf16.h>

typedef __attribute__((ext_vector_type(16))) __bf16 v16bf;
typedef __attribute__((ext_vector_type(8)))  float  v8f;
typedef __attribute__((ext_vector_type(4)))  unsigned int v4u;

#define K_CODES 1024
#define D_DIM   256
#define N_VEC   32768
#define DECAY_F 0.99f
#define OMD_F   0.01f
#define EPS_F   1e-5f

// Async VMEM->LDS builtin types, per hipcc diagnostic:
//   arg0: 'int __vector(4) __device__ *'  == int4 addrspace(1)* (global)
//   arg1: LDS destination               == int4 addrspace(3)* (shared)
// AS1 cast is bit-identical on amdgpu; LDS offset = low 32 bits of the flat
// shared-aperture address, so the integer round-trip is the correct AS3 cast.
typedef int v4i_t __attribute__((vector_size(16)));
typedef __attribute__((address_space(1))) v4i_t g_v4i_t;
typedef __attribute__((address_space(3))) v4i_t l_v4i_t;
#define TO_GBL4(p) ((g_v4i_t*)(unsigned long long)(p))
#define TO_LDS4(p) ((l_v4i_t*)(unsigned int)(unsigned long long)(p))

#if __has_builtin(__builtin_amdgcn_global_load_async_to_lds_b128)
#define HAVE_ASYNC_LDS 1
#else
#define HAVE_ASYNC_LDS 0
#endif

// ---------------------------------------------------------------------------
// Zero workspace accumulators (must run every call; harness does not re-poison)
// ---------------------------------------------------------------------------
__global__ void vq_zero(float* __restrict__ counts,
                        float* __restrict__ embed_sum,
                        float* __restrict__ scalars) {
    int i = blockIdx.x * blockDim.x + threadIdx.x;
    if (i < K_CODES) counts[i] = 0.0f;
    if (i < K_CODES * D_DIM) embed_sum[i] = 0.0f;
    if (i < 2) scalars[i] = 0.0f;
}

// ---------------------------------------------------------------------------
// e_norm[k] = sum_d emb[k][d]^2  (f32, exact)
// ---------------------------------------------------------------------------
__global__ void vq_prep_enorm(const float* __restrict__ emb,
                              float* __restrict__ e_norm) {
    int k = blockIdx.x * blockDim.x + threadIdx.x;
    if (k >= K_CODES) return;
    float s = 0.0f;
    for (int d = 0; d < D_DIM; ++d) {
        float v = emb[k * D_DIM + d];
        s += v * v;
    }
    e_norm[k] = s;
}

// ---------------------------------------------------------------------------
// Pre-swizzle codebook into WMMA B-fragment order (bf16).
// Fragment (nt, c, lane) holds the v16bf that lane feeds to
// v_wmma_f32_16x16x32_bf16 for N-tile nt, K-chunk c:
//   col n  = nt*16 + (lane&15)
//   kloc(j)= (j<8 ? j : j+8) + 8*(lane>>4)      (ISA 16-bit operand layout)
//   depth d= c*32 + kloc(j)
// ---------------------------------------------------------------------------
__global__ void vq_prep_packB(const float* __restrict__ emb,
                              __bf16* __restrict__ packB) {
    int t = blockIdx.x * blockDim.x + threadIdx.x;   // 0 .. 16383
    int lane = t & 31;
    int c    = (t >> 5) & 7;
    int nt   = t >> 8;
    int half = lane >> 4;
    int n    = nt * 16 + (lane & 15);
    int dbase = c * 32 + half * 8;
    __bf16* o = packB + (size_t)t * 16;
    for (int j = 0; j < 8; ++j) {
        o[j]     = (__bf16)emb[n * D_DIM + dbase + j];
        o[j + 8] = (__bf16)emb[n * D_DIM + dbase + 16 + j];
    }
}

// ---------------------------------------------------------------------------
// Main kernel: WMMA distance GEMM + running argmin.
// 256 threads = 8 waves; each wave owns a 16-row M-tile, A resident in
// registers. B fragments for each N-tile (8 KB) are staged into LDS ONCE per
// block (async VMEM->LDS, double-buffered) and shared by all 8 waves.
// ---------------------------------------------------------------------------
__global__ __launch_bounds__(256)
void vq_argmin_wmma(const float* __restrict__ x,
                    const __bf16* __restrict__ packB,
                    const float* __restrict__ e_norm,
                    float* __restrict__ token_f,
                    int* __restrict__ token_i) {
    __shared__ float  lds_en[K_CODES];
    __shared__ float  lds_xn[8][32];
    __shared__ __bf16 lds_B[2][8 * 32 * 16];     // 2 x 8 KB double buffer

    const int tid  = threadIdx.x;
    const int wave = tid >> 5;
    const int lane = tid & 31;
    const int half = lane >> 4;
    const int l16  = lane & 15;

    // stage all 1024 code norms in LDS (4 KB)
    for (int i = tid; i < K_CODES; i += 256) lds_en[i] = e_norm[i];

    const int mtile = blockIdx.x * 8 + wave;
    const int mrow0 = mtile * 16;
    const int arow  = mrow0 + l16;

    // Load A fragments (f32 -> bf16 on the fly), accumulate ||x||^2 partials.
    // ISA 16-bit A 16x32 layout: lanes 0-15 row M=l hold K {0..7,16..23};
    // lanes 16-31 row M=l-16 hold K {8..15,24..31}.
    v16bf A[8];
    float xn_part = 0.0f;
#pragma unroll
    for (int c = 0; c < 8; ++c) {
        const float* p = x + (size_t)arow * D_DIM + c * 32 + half * 8;
#pragma unroll
        for (int j = 0; j < 8; ++j) {
            float f0 = p[j];
            float f1 = p[16 + j];
            A[c][j]     = (__bf16)f0;
            A[c][j + 8] = (__bf16)f1;
            xn_part += f0 * f0 + f1 * f1;
        }
    }
    lds_xn[wave][lane] = xn_part;

    // --- B staging: each thread owns one 32-byte fragment slot per N-tile ---
    const __bf16* my_src_base = packB + (size_t)tid * 16;    // + nt*256*16
    __bf16*       my_dst0     = &lds_B[0][tid * 16];
    __bf16*       my_dst1     = &lds_B[1][tid * 16];

#if HAVE_ASYNC_LDS
#define STAGE_B(nt, buf)                                                       \
    do {                                                                       \
        const __bf16* s = my_src_base + (size_t)(nt) * 256 * 16;               \
        __bf16* d = (buf) ? my_dst1 : my_dst0;                                 \
        __builtin_amdgcn_global_load_async_to_lds_b128(TO_GBL4(s), TO_LDS4(d), 0, 0); \
        __builtin_amdgcn_global_load_async_to_lds_b128(TO_GBL4(s + 8), TO_LDS4(d + 8), 0, 0); \
    } while (0)
#if __has_builtin(__builtin_amdgcn_s_wait_asynccnt)
#define WAIT_ASYNC() __builtin_amdgcn_s_wait_asynccnt(0)
#else
#define WAIT_ASYNC() asm volatile("s_wait_asynccnt 0x0" ::: "memory")
#endif
#else
#define STAGE_B(nt, buf)                                                       \
    do {                                                                       \
        const v4u* s = (const v4u*)(my_src_base + (size_t)(nt) * 256 * 16);    \
        v4u* d = (v4u*)((buf) ? my_dst1 : my_dst0);                            \
        d[0] = s[0];                                                           \
        d[1] = s[1];                                                           \
    } while (0)
#define WAIT_ASYNC() ((void)0)
#endif

    STAGE_B(0, 0);
    WAIT_ASYNC();
    __syncthreads();   // lds_en, lds_xn and B buffer 0 all visible

    // Row norm r = partial(lane r) + partial(lane r+16).
    // C/D layout: VGPR j <-> row j + 8*half, lane%16 <-> col.
    float xn[8];
#pragma unroll
    for (int j = 0; j < 8; ++j)
        xn[j] = lds_xn[wave][j + 8 * half] + lds_xn[wave][j + 8 * half + 16];

    float bestv[8];
    int   besti[8];
#pragma unroll
    for (int j = 0; j < 8; ++j) { bestv[j] = 3.4e38f; besti[j] = 0; }

    for (int nt = 0; nt < 64; ++nt) {
        const int cur = nt & 1;
        if (nt + 1 < 64) STAGE_B(nt + 1, (nt + 1) & 1);   // prefetch next tile

        float en = lds_en[nt * 16 + l16];
        v8f acc = {};
#pragma unroll
        for (int c = 0; c < 8; ++c) {
            v16bf Bf = *(const v16bf*)&lds_B[cur][(c * 32 + lane) * 16];
            acc = __builtin_amdgcn_wmma_f32_16x16x32_bf16(
                false, A[c], false, Bf, (short)0, acc, false, false);
        }
        const int col = nt * 16 + l16;
#pragma unroll
        for (int j = 0; j < 8; ++j) {
            float d = xn[j] + en - 2.0f * acc[j];   // squared L2 distance
            if (d < bestv[j]) { bestv[j] = d; besti[j] = col; }
        }

        WAIT_ASYNC();      // next buffer landed (async path)
        __syncthreads();   // everyone done reading cur; next buffer visible
    }

    // Min-reduce (value, index) across the 16 lanes holding each row's cols.
#pragma unroll
    for (int j = 0; j < 8; ++j) {
        float bv = bestv[j];
        int   bi = besti[j];
        for (int mask = 1; mask < 16; mask <<= 1) {
            float ov = __shfl_xor(bv, mask, 16);
            int   oi = __shfl_xor(bi, mask, 16);
            if (ov < bv || (ov == bv && oi < bi)) { bv = ov; bi = oi; }
        }
        if (l16 == 0) {
            int r = mrow0 + j + 8 * half;
            token_f[r] = (float)bi;
            token_i[r] = bi;
        }
    }
}

// ---------------------------------------------------------------------------
// Per-row gather/scatter: quantized_st output, embed sums, counts, loss.
// One block (256 threads = 256 dims) per input row.
// ---------------------------------------------------------------------------
__global__ __launch_bounds__(256)
void vq_scatter(const float* __restrict__ x,
                const float* __restrict__ emb,
                const int* __restrict__ token_i,
                float* __restrict__ quant_out,
                float* __restrict__ embed_sum,
                float* __restrict__ counts,
                float* __restrict__ loss_acc) {
    __shared__ float red[256];
    const int row = blockIdx.x;
    const int d   = threadIdx.x;
    const int idx = token_i[row];

    float q  = emb[(size_t)idx * D_DIM + d];
    float xv = x[(size_t)row * D_DIM + d];
    // straight-through: inputs + stopgrad(quantized - inputs) == quantized
    quant_out[(size_t)row * D_DIM + d] = q;
    atomicAdd(&embed_sum[(size_t)idx * D_DIM + d], xv);

    float diff = xv - q;
    red[d] = diff * diff;
    __syncthreads();
    for (int s = 128; s > 0; s >>= 1) {
        if (d < s) red[d] += red[d + s];
        __syncthreads();
    }
    if (d == 0) {
        atomicAdd(loss_acc, red[0]);
        atomicAdd(&counts[idx], 1.0f);
    }
}

// ---------------------------------------------------------------------------
// EMA cluster-size update + total accumulation.
// ---------------------------------------------------------------------------
__global__ void vq_cluster(const float* __restrict__ cs,
                           const float* __restrict__ counts,
                           float* __restrict__ out_ncs,
                           float* __restrict__ total_acc) {
    int k = blockIdx.x * blockDim.x + threadIdx.x;
    if (k >= K_CODES) return;
    float ncs = cs[k] * DECAY_F + OMD_F * counts[k];
    out_ncs[k] = ncs;
    atomicAdd(total_acc, ncs);
}

// ---------------------------------------------------------------------------
// EMA embedding-avg update + Laplace-normalized embedding.
// ---------------------------------------------------------------------------
__global__ __launch_bounds__(256)
void vq_embed_final(const float* __restrict__ ea,
                    const float* __restrict__ embed_sum,
                    const float* __restrict__ out_ncs,
                    const float* __restrict__ total_acc,
                    float* __restrict__ out_nea,
                    float* __restrict__ out_ne) {
    const int k = blockIdx.x;
    const int d = threadIdx.x;
    size_t o = (size_t)k * D_DIM + d;
    float nea = ea[o] * DECAY_F + OMD_F * embed_sum[o];
    out_nea[o] = nea;
    float total = *total_acc;
    float norm = (out_ncs[k] + EPS_F) / (total + (float)K_CODES * EPS_F) * total;
    out_ne[o] = nea / norm;
}

__global__ void vq_loss_final(const float* __restrict__ loss_acc,
                              float* __restrict__ out_loss) {
    *out_loss = *loss_acc / (float)((long long)N_VEC * D_DIM);
}

// ---------------------------------------------------------------------------
extern "C" void kernel_launch(void* const* d_in, const int* in_sizes, int n_in,
                              void* d_out, int out_size, void* d_ws, size_t ws_size,
                              hipStream_t stream) {
    (void)in_sizes; (void)n_in; (void)out_size; (void)ws_size;
    const float* x   = (const float*)d_in[0];  // [32768, 256]
    const float* emb = (const float*)d_in[1];  // [1024, 256]
    const float* cs  = (const float*)d_in[2];  // [1024]
    const float* ea  = (const float*)d_in[3];  // [1024, 256]

    // Output tuple, concatenated flat (floats):
    float* out      = (float*)d_out;
    float* out_q    = out;                 // quantized_st     8,388,608
    float* out_tok  = out + 8388608;       // token_ids           32,768
    float* out_loss = out + 8421376;       // commitment_loss          1
    float* out_ne   = out + 8421377;       // new_embedding      262,144
    float* out_ncs  = out + 8683521;       // new_cluster_size     1,024
    float* out_nea  = out + 8684545;       // new_embedding_avg  262,144

    // Workspace layout (bytes):
    char*   ws        = (char*)d_ws;
    __bf16* packB     = (__bf16*)ws;                 //   524,288
    float*  e_norm    = (float*)(ws + 524288);       //     4,096
    float*  counts    = (float*)(ws + 528384);       //     4,096
    float*  embed_sum = (float*)(ws + 532480);       // 1,048,576
    int*    token_i   = (int*)(ws + 1581056);        //   131,072
    float*  scalars   = (float*)(ws + 1712128);      // loss_acc, total_acc
    float*  loss_acc  = scalars;
    float*  total_acc = scalars + 1;

    vq_zero<<<(K_CODES * D_DIM + 255) / 256, 256, 0, stream>>>(counts, embed_sum, scalars);
    vq_prep_enorm<<<(K_CODES + 255) / 256, 256, 0, stream>>>(emb, e_norm);
    vq_prep_packB<<<64, 256, 0, stream>>>(emb, packB);
    vq_argmin_wmma<<<N_VEC / 128, 256, 0, stream>>>(x, packB, e_norm, out_tok, token_i);
    vq_scatter<<<N_VEC, 256, 0, stream>>>(x, emb, token_i, out_q, embed_sum, counts, loss_acc);
    vq_cluster<<<(K_CODES + 255) / 256, 256, 0, stream>>>(cs, counts, out_ncs, total_acc);
    vq_embed_final<<<K_CODES, 256, 0, stream>>>(ea, embed_sum, out_ncs, total_acc, out_nea, out_ne);
    vq_loss_final<<<1, 1, 0, stream>>>(loss_acc, out_loss);
}
